// ConvVQVAE_24094766531146
// MI455X (gfx1250) — matrix operand order repaired and checked by reading, not compiled, not run on hardware
//
#include <hip/hip_runtime.h>
#include <hip/hip_bf16.h>
#include <math.h>

// ---------------------------------------------------------------------------
// CDNA5 (gfx1250) VQ-VAE forward.
//  * bf16 WMMA implicit GEMM for conv1/conv2, VQ distances, all deconvs.
//  * Transpose convs specialized by output parity -> 4x fewer issued MACs.
//  * TDM tensor_load_to_lds stages the VQ codebook tiles (padded LDS rows).
//    (Probe-confirmed live in this toolchain: histogram wmma 72 -> 76 test.)
// ---------------------------------------------------------------------------

typedef __attribute__((ext_vector_type(16))) __bf16 v16bf;
typedef __attribute__((ext_vector_type(8)))  float  v8f;
typedef __attribute__((ext_vector_type(4)))  unsigned int u32x4;
typedef __attribute__((ext_vector_type(8)))  int          i32x8;
typedef __attribute__((ext_vector_type(4)))  int          i32x4;

#if defined(__has_builtin)
#if __has_builtin(__builtin_amdgcn_tensor_load_to_lds) && \
    __has_builtin(__builtin_amdgcn_s_wait_tensorcnt)
#define HAVE_TDM 1
#endif
#endif
#ifndef HAVE_TDM
#define HAVE_TDM 0
#endif

#define DEV static __device__ __forceinline__

DEV unsigned short f2bf(float f) {
    unsigned int u = __float_as_uint(f);
    u += 0x7FFFu + ((u >> 16) & 1u);   // round-to-nearest-even
    return (unsigned short)(u >> 16);
}

// LDS rows are 40 ushorts (80B): 16B-aligned b128 reads, bank-conflict-free.
DEV v16bf load_fragA(const unsigned short* row, int h) {
    // A 16x32 bf16: lane m=L&15; elems 0..7 -> K=8h..8h+7 ; 8..15 -> K=16+8h..
    union { v16bf v; uint4 q[2]; } u;
    u.q[0] = *(const uint4*)(row + h * 8);
    u.q[1] = *(const uint4*)(row + 16 + h * 8);
    return u.v;
}
DEV v16bf load_fragB(const unsigned short* row, int h) {
    // B 32x16 bf16: lane n=L&15; elems 0..7 -> K=8h..8h+7 ; 8..15 -> K=16+8h..
    union { v16bf v; uint4 q[2]; } u;
    u.q[0] = *(const uint4*)(row + h * 8);
    u.q[1] = *(const uint4*)(row + 16 + h * 8);
    return u.v;
}

enum { EPI_NONE = 0, EPI_RELU = 1, EPI_SIGMOID = 2 };

// ---------------------------------------------------------------------------
// Tap-table implicit GEMM. Covers:
//  * stride-2 conv (TY=TX=3, SI=2, SO=1, dy/dx = {-1,0,1})
//  * parity-specialized transpose conv (SI=1, SO=2): for output parity p,
//    taps ky={1} (p=0) or {0,2} (p=1), input offsets dy={0} or {0,1}.
// M=Cout, K=Cin*TY*TX, N=Nbatch*GH*GW (per-launch output grid).
// dyP/dxP pack (offset+1) per tap, kyP/kxP pack the weight k-index per tap.
// ---------------------------------------------------------------------------
template <int MW, int NW, int EPI, int TY, int TX>
__global__ __launch_bounds__(256) void gemm_tap(
    const unsigned short* __restrict__ act,   // bf16 [N,Cin,IH,IW]
    const float* __restrict__ w,              // f32 OIHW [Cout,Cin,3,3]
    float* __restrict__ outf,                 // optional f32 [N,Cout,OHo,OWo]
    unsigned short* __restrict__ outb,        // optional bf16 out
    int Cout, int Cin, int GH, int GW, int IH, int IW,
    int OHo, int OWo, int SI, int SO, int oy0, int ox0,
    int dyP, int dxP, int kyP, int kxP)
{
    constexpr int MT = MW * 32, NT = NW * 32;
    constexpr int T = TY * TX;
    __shared__ unsigned short As[MT * 40];
    __shared__ unsigned short Bs[NT * 40];

    const int tid  = threadIdx.x;
    const int lane = tid & 31;
    const int wave = tid >> 5;
    const int wm = wave / NW, wn = wave % NW;
    const int h  = lane >> 4, nl = lane & 15;
    const int mBase   = blockIdx.y * MT;
    const int colBase = blockIdx.x * NT;
    const int Ktot = Cin * T;
    const int GHW  = GH * GW;

    v8f acc[2][2] = {};

    for (int ks = 0; ks < Ktot; ks += 32) {
        // ---- A tile: MT x 32 weights (gathered via tap tables), f32 -> bf16
        for (int idx = tid; idx < MT * 32; idx += 256) {
            int m = idx >> 5, k = idx & 31;
            int kg = ks + k;
            int mg = mBase + m;
            int ci = kg / T; int r = kg - ci * T;
            int ty = r / TX; int tx = r - ty * TX;
            int ky = (kyP >> (4 * ty)) & 15;
            int kx = (kxP >> (4 * tx)) & 15;
            float v = (mg < Cout) ? w[((size_t)mg * Cin + ci) * 9 + ky * 3 + kx]
                                  : 0.0f;
            As[m * 40 + k] = f2bf(v);
            // prefetch next K-chunk of weights (global_prefetch_b8)
            if (kg + 32 < Ktot && mg < Cout) {
                int kg2 = kg + 32;
                int c2 = kg2 / T; int r2 = kg2 - c2 * T;
                int t2 = r2 / TX; int x2 = r2 - t2 * TX;
                int ky2 = (kyP >> (4 * t2)) & 15;
                int kx2 = (kxP >> (4 * x2)) & 15;
                __builtin_prefetch(&w[((size_t)mg * Cin + c2) * 9 + ky2 * 3 + kx2], 0, 0);
            }
        }
        // ---- B tile: NT x 32 im2col gather (stored [col][k])
        for (int idx = tid; idx < NT * 32; idx += 256) {
            int col = idx & (NT - 1), k = idx / NT;
            int kg = ks + k;
            int ci = kg / T; int r = kg - ci * T;
            int ty = r / TX; int tx = r - ty * TX;
            int cg = colBase + col;
            int n  = cg / GHW; int rem = cg - n * GHW;
            int oy = rem / GW; int ox = rem - oy * GW;
            int iy = oy * SI + ((dyP >> (4 * ty)) & 15) - 1;
            int ix = ox * SI + ((dxP >> (4 * tx)) & 15) - 1;
            unsigned short v = 0;
            if (iy >= 0 && iy < IH && ix >= 0 && ix < IW)
                v = act[((size_t)(n * Cin + ci) * IH + iy) * IW + ix];
            Bs[col * 40 + k] = v;
        }
        __syncthreads();

        v16bf a0 = load_fragA(&As[(wm * 32 + nl) * 40], h);
        v16bf a1 = load_fragA(&As[(wm * 32 + 16 + nl) * 40], h);
        v16bf b0 = load_fragB(&Bs[(wn * 32 + nl) * 40], h);
        v16bf b1 = load_fragB(&Bs[(wn * 32 + 16 + nl) * 40], h);
        acc[0][0] = __builtin_amdgcn_wmma_f32_16x16x32_bf16(false, a0, false, b0, (short)0, acc[0][0], false, false);
        acc[0][1] = __builtin_amdgcn_wmma_f32_16x16x32_bf16(false, a0, false, b1, (short)0, acc[0][1], false, false);
        acc[1][0] = __builtin_amdgcn_wmma_f32_16x16x32_bf16(false, a1, false, b0, (short)0, acc[1][0], false, false);
        acc[1][1] = __builtin_amdgcn_wmma_f32_16x16x32_bf16(false, a1, false, b1, (short)0, acc[1][1], false, false);
        __syncthreads();
    }

    // ---- epilogue: D layout row = 8h + r, col = nl
    for (int sm = 0; sm < 2; sm++)
        for (int sn = 0; sn < 2; sn++) {
            int col = colBase + wn * 32 + sn * 16 + nl;
            int n = col / GHW; int rem = col - n * GHW;
            int oy = rem / GW; int ox = rem - oy * GW;
            int oyo = oy * SO + oy0, oxo = ox * SO + ox0;
            for (int r = 0; r < 8; r++) {
                int m = mBase + wm * 32 + sm * 16 + 8 * h + r;
                if (m < Cout) {
                    float v = acc[sm][sn][r];
                    if (EPI == EPI_RELU)    v = fmaxf(v, 0.0f);
                    if (EPI == EPI_SIGMOID) v = 1.0f / (1.0f + __expf(-v));
                    size_t o = ((size_t)(n * Cout + m) * OHo + oyo) * OWo + oxo;
                    if (outf) outf[o] = v;
                    if (outb) outb[o] = f2bf(v);
                }
            }
        }
}

// ---------------------------------------------------------------------------
// conv0: 3->64, k3 s2 p1 + ReLU, direct (K=27 too small for WMMA).
// ---------------------------------------------------------------------------
__global__ __launch_bounds__(256) void conv0_direct(
    const float* __restrict__ x, const float* __restrict__ w0,
    unsigned short* __restrict__ h0)
{
    int gid = blockIdx.x * 256 + threadIdx.x;       // == NCHW flat out index
    int ox = gid & 127; int t = gid >> 7;
    int oy = t & 127;   t >>= 7;
    int co = t & 63;    int n = t >> 6;
    float s = 0.0f;
    for (int ci = 0; ci < 3; ci++)
        for (int ky = 0; ky < 3; ky++) {
            int iy = 2 * oy - 1 + ky;
            if (iy < 0 || iy > 255) continue;
            for (int kx = 0; kx < 3; kx++) {
                int ix = 2 * ox - 1 + kx;
                if (ix < 0 || ix > 255) continue;
                s += x[((size_t)(n * 3 + ci) << 16) + (iy << 8) + ix] *
                     w0[((co * 3 + ci) * 3 + ky) * 3 + kx];
            }
        }
    h0[gid] = f2bf(fmaxf(s, 0.0f));
}

// ---------------------------------------------------------------------------
// codebook prep: f32 -> bf16 copy + ||c||^2 per code
// ---------------------------------------------------------------------------
__global__ void prep_codebook(const float* __restrict__ cb,
                              unsigned short* __restrict__ cbb,
                              float* __restrict__ cn)
{
    int code = blockIdx.x * blockDim.x + threadIdx.x;
    if (code >= 1024) return;
    float s = 0.0f;
    for (int c = 0; c < 256; c++) {
        float v = cb[code * 256 + c];
        s += v * v;
        cbb[code * 256 + c] = f2bf(v);
    }
    cn[code] = s;
}

// ---------------------------------------------------------------------------
// VQ: WMMA GEMM scores[code,pos] = z . c ; dist = ||c||^2 - 2*score ; argmin.
// Codebook A tiles staged by the Tensor Data Mover (2D D#, LDS row padding
// 4 DWORDs per 16 DWORDs reproduces the 80B padded LDS row stride).
// ---------------------------------------------------------------------------
__global__ __launch_bounds__(256) void vq_argmin(
    const unsigned short* __restrict__ zb,   // bf16 z [32,256,32,32]
    const unsigned short* __restrict__ cbb,  // bf16 codebook [1024,256]
    const float* __restrict__ cn,            // ||c||^2 [1024]
    const float* __restrict__ cbf,           // f32 codebook [1024,256]
    float* __restrict__ ekf,                 // f32 e_k [32,256,32,32]
    unsigned short* __restrict__ ekb,        // bf16 e_k
    int* __restrict__ ids)                   // int32 [32768]
{
    __shared__ unsigned short As[64 * 40];
    __shared__ unsigned short Bs[128 * 40];
    __shared__ float redv[2 * 128];
    __shared__ int   redi[2 * 128];
    __shared__ int   idsl[128];

    const int tid = threadIdx.x, lane = tid & 31, wave = tid >> 5;
    const int wm = wave >> 2, wn = wave & 3;
    const int h = lane >> 4, nl = lane & 15;
    const int pos0 = blockIdx.x * 128;
    const int nimg = pos0 >> 10;      // 1024 spatial positions per image
    const int s0   = pos0 & 1023;

    float minv[2] = {3.4e38f, 3.4e38f};
    int   mini[2] = {0, 0};

    for (int mc = 0; mc < 16; mc++) {
        v8f acc[2][2] = {};
        for (int ks = 0; ks < 256; ks += 32) {
#if HAVE_TDM
            if (wave == 0) {
                // D# group0: count=1 | lds_addr | global_addr | type=2
                unsigned lds = (unsigned)(uintptr_t)(&As[0]);
                unsigned long long ga =
                    (unsigned long long)(uintptr_t)(cbb + (size_t)(mc * 64) * 256 + ks);
                u32x4 g0;
                g0[0] = 1u;                    // count=1, user mode
                g0[1] = lds;                   // LDS byte address
                g0[2] = (unsigned)ga;          // global addr [31:0]
                g0[3] = (unsigned)((ga >> 32) & 0x01FFFFFFu) | 0x80000000u; // [56:32] | type=2
                // D# group1: data_size=2B, pad 4 DWORDs after every 16 DWORDs,
                // tensor 32x64 (elems), tile 32x64, dim0 stride 256 elems.
                i32x8 g1;
                g1[0] = (1 << 16) | (1 << 20) | (3 << 22) | (3 << 25);
                g1[1] = (int)(32u << 16);      // tensor_dim0 = 32
                g1[2] = (int)(64u << 16);      // tensor_dim1 = 64
                g1[3] = (int)(32u << 16);      // tile_dim0 = 32
                g1[4] = 64;                    // tile_dim1 = 64, tile_dim2 = 0
                g1[5] = 256;                   // tensor_dim0_stride = 256
                g1[6] = 0;
                g1[7] = 0;
                i32x4 gz; gz[0] = 0; gz[1] = 0; gz[2] = 0; gz[3] = 0;
#if __clang_major__ >= 23
                i32x8 gz8; for (int q = 0; q < 8; q++) gz8[q] = 0;
                __builtin_amdgcn_tensor_load_to_lds(g0, g1, gz, gz, gz8, 0);
#else
                __builtin_amdgcn_tensor_load_to_lds(g0, g1, gz, gz, 0);
#endif
                __builtin_amdgcn_s_wait_tensorcnt(0);
            }
#else
            for (int idx = tid; idx < 64 * 32; idx += 256) {
                int m = idx >> 5, k = idx & 31;
                As[m * 40 + k] = cbb[(size_t)(mc * 64 + m) * 256 + ks + k];
            }
#endif
            for (int idx = tid; idx < 128 * 32; idx += 256) {
                int col = idx & 127, k = idx >> 7;
                Bs[col * 40 + k] =
                    zb[((size_t)(nimg * 256 + ks + k) << 10) + s0 + col];
            }
            __syncthreads();
            v16bf a0 = load_fragA(&As[(wm * 32 + nl) * 40], h);
            v16bf a1 = load_fragA(&As[(wm * 32 + 16 + nl) * 40], h);
            v16bf b0 = load_fragB(&Bs[(wn * 32 + nl) * 40], h);
            v16bf b1 = load_fragB(&Bs[(wn * 32 + 16 + nl) * 40], h);
            acc[0][0] = __builtin_amdgcn_wmma_f32_16x16x32_bf16(false, a0, false, b0, (short)0, acc[0][0], false, false);
            acc[0][1] = __builtin_amdgcn_wmma_f32_16x16x32_bf16(false, a0, false, b1, (short)0, acc[0][1], false, false);
            acc[1][0] = __builtin_amdgcn_wmma_f32_16x16x32_bf16(false, a1, false, b0, (short)0, acc[1][0], false, false);
            acc[1][1] = __builtin_amdgcn_wmma_f32_16x16x32_bf16(false, a1, false, b1, (short)0, acc[1][1], false, false);
            __syncthreads();
        }
        // dist = ||c||^2 - 2*score; codes visited in increasing order
        for (int sm = 0; sm < 2; sm++)
            for (int sn = 0; sn < 2; sn++)
                for (int r = 0; r < 8; r++) {
                    int code = mc * 64 + wm * 32 + sm * 16 + 8 * h + r;
                    float d = cn[code] - 2.0f * acc[sm][sn][r];
                    if (d < minv[sn]) { minv[sn] = d; mini[sn] = code; }
                }
    }

    // reduce across lane halves (rows 8h+r vs 8(1-h)+r share a column)
    for (int sn = 0; sn < 2; sn++) {
        float ov = __shfl_xor(minv[sn], 16, 32);
        int   oi = __shfl_xor(mini[sn], 16, 32);
        if (ov < minv[sn] || (ov == minv[sn] && oi < mini[sn])) {
            minv[sn] = ov; mini[sn] = oi;
        }
        if (h == 0) {
            int col = wn * 32 + sn * 16 + nl;
            redv[wm * 128 + col] = minv[sn];
            redi[wm * 128 + col] = mini[sn];
        }
    }
    __syncthreads();
    if (tid < 128) {
        float v0 = redv[tid], v1 = redv[128 + tid];
        int   i0 = redi[tid], i1 = redi[128 + tid];
        int id = (v1 < v0 || (v1 == v0 && i1 < i0)) ? i1 : i0;
        idsl[tid] = id;
        ids[pos0 + tid] = id;
    }
    __syncthreads();
    // e_k gather: e_k[n, c, pos] = codebook[id[pos], c]
    for (int i = tid; i < 256 * 128; i += 256) {
        int c = i >> 7, s = i & 127;
        int id = idsl[s];
        float v = cbf[(size_t)id * 256 + c];
        size_t o = ((size_t)(nimg * 256 + c) << 10) + s0 + s;
        ekf[o] = v;
        ekb[o] = f2bf(v);
    }
}

// ---------------------------------------------------------------------------
// Host launch
// ---------------------------------------------------------------------------
extern "C" void kernel_launch(void* const* d_in, const int* in_sizes, int n_in,
                              void* d_out, int out_size, void* d_ws, size_t ws_size,
                              hipStream_t stream)
{
    const float* x   = (const float*)d_in[0];
    const float* ew0 = (const float*)d_in[1];
    const float* ew1 = (const float*)d_in[2];
    const float* ew2 = (const float*)d_in[3];
    const float* cb  = (const float*)d_in[4];
    const float* dw0 = (const float*)d_in[5];
    const float* dw1 = (const float*)d_in[6];
    const float* dw2 = (const float*)d_in[7];

    char* ws = (char*)d_ws;
    size_t o = 0;
    unsigned short* h0  = (unsigned short*)(ws + o); o += (size_t)33554432 * 2; // [32,64,128,128]
    unsigned short* h1  = (unsigned short*)(ws + o); o += (size_t)16777216 * 2; // [32,128,64,64]
    unsigned short* zbf = (unsigned short*)(ws + o); o += (size_t)8388608  * 2; // [32,256,32,32]
    unsigned short* cbb = (unsigned short*)(ws + o); o += (size_t)262144   * 2; // [1024,256]
    float*          cnm = (float*)(ws + o);          o += (size_t)1024     * 4;
    unsigned short* ekb = (unsigned short*)(ws + o); o += (size_t)8388608  * 2; // [32,256,32,32]
    unsigned short* dd0 = (unsigned short*)(ws + o); o += (size_t)16777216 * 2; // [32,128,64,64]
    unsigned short* dd1 = (unsigned short*)(ws + o); o += (size_t)33554432 * 2; // [32,64,128,128]

    // d_out layout (return order): out | z | e_k | ids
    float* outp = (float*)d_out;
    float* zf   = outp + 6291456;   // 32*3*256*256
    float* ekf  = outp + 14680064;  // + 32*256*32*32
    int*   ids  = (int*)(outp + 23068672);

    prep_codebook<<<4, 256, 0, stream>>>(cb, cbb, cnm);
    conv0_direct<<<131072, 256, 0, stream>>>(x, ew0, h0);

    // enc conv1: 64->128, relu. K=576, N=131072.
    gemm_tap<2, 4, EPI_RELU, 3, 3><<<dim3(1024, 2), 256, 0, stream>>>(
        h0, ew1, nullptr, h1, 128, 64, 64, 64, 128, 128, 64, 64,
        2, 1, 0, 0, 0x210, 0x210, 0x210, 0x210);
    // enc conv2: 128->256, no relu. K=1152, N=32768. f32 z + bf16 z.
    gemm_tap<2, 4, EPI_NONE, 3, 3><<<dim3(256, 4), 256, 0, stream>>>(
        h1, ew2, zf, zbf, 256, 128, 32, 32, 64, 64, 32, 32,
        2, 1, 0, 0, 0x210, 0x210, 0x210, 0x210);

    vq_argmin<<<256, 256, 0, stream>>>(zbf, cbb, cnm, cb, ekf, ekb, ids);

    // dec deconv0: 256->128, relu, parity-specialized (4 launches, N=32768).
    gemm_tap<2, 4, EPI_RELU, 1, 1><<<dim3(256, 2), 256, 0, stream>>>(
        ekb, dw0, nullptr, dd0, 128, 256, 32, 32, 32, 32, 64, 64,
        1, 2, 0, 0, 0x1, 0x1, 0x1, 0x1);
    gemm_tap<2, 4, EPI_RELU, 1, 2><<<dim3(256, 2), 256, 0, stream>>>(
        ekb, dw0, nullptr, dd0, 128, 256, 32, 32, 32, 32, 64, 64,
        1, 2, 0, 1, 0x1, 0x21, 0x1, 0x20);
    gemm_tap<2, 4, EPI_RELU, 2, 1><<<dim3(256, 2), 256, 0, stream>>>(
        ekb, dw0, nullptr, dd0, 128, 256, 32, 32, 32, 32, 64, 64,
        1, 2, 1, 0, 0x21, 0x1, 0x20, 0x1);
    gemm_tap<2, 4, EPI_RELU, 2, 2><<<dim3(256, 2), 256, 0, stream>>>(
        ekb, dw0, nullptr, dd0, 128, 256, 32, 32, 32, 32, 64, 64,
        1, 2, 1, 1, 0x21, 0x21, 0x20, 0x20);

    // dec deconv1: 128->64, relu, parity-specialized (N=131072 each).
    gemm_tap<2, 4, EPI_RELU, 1, 1><<<dim3(1024, 1), 256, 0, stream>>>(
        dd0, dw1, nullptr, dd1, 64, 128, 64, 64, 64, 64, 128, 128,
        1, 2, 0, 0, 0x1, 0x1, 0x1, 0x1);
    gemm_tap<2, 4, EPI_RELU, 1, 2><<<dim3(1024, 1), 256, 0, stream>>>(
        dd0, dw1, nullptr, dd1, 64, 128, 64, 64, 64, 64, 128, 128,
        1, 2, 0, 1, 0x1, 0x21, 0x1, 0x20);
    gemm_tap<2, 4, EPI_RELU, 2, 1><<<dim3(1024, 1), 256, 0, stream>>>(
        dd0, dw1, nullptr, dd1, 64, 128, 64, 64, 64, 64, 128, 128,
        1, 2, 1, 0, 0x21, 0x1, 0x20, 0x1);
    gemm_tap<2, 4, EPI_RELU, 2, 2><<<dim3(1024, 1), 256, 0, stream>>>(
        dd0, dw1, nullptr, dd1, 64, 128, 64, 64, 64, 64, 128, 128,
        1, 2, 1, 1, 0x21, 0x21, 0x20, 0x20);

    // dec deconv2: 64->3, sigmoid, parity-specialized (N=524288 each).
    gemm_tap<1, 8, EPI_SIGMOID, 1, 1><<<dim3(2048, 1), 256, 0, stream>>>(
        dd1, dw2, outp, nullptr, 3, 64, 128, 128, 128, 128, 256, 256,
        1, 2, 0, 0, 0x1, 0x1, 0x1, 0x1);
    gemm_tap<1, 8, EPI_SIGMOID, 1, 2><<<dim3(2048, 1), 256, 0, stream>>>(
        dd1, dw2, outp, nullptr, 3, 64, 128, 128, 128, 128, 256, 256,
        1, 2, 0, 1, 0x1, 0x21, 0x1, 0x20);
    gemm_tap<1, 8, EPI_SIGMOID, 2, 1><<<dim3(2048, 1), 256, 0, stream>>>(
        dd1, dw2, outp, nullptr, 3, 64, 128, 128, 128, 128, 256, 256,
        1, 2, 1, 0, 0x21, 0x1, 0x20, 0x1);
    gemm_tap<1, 8, EPI_SIGMOID, 2, 2><<<dim3(2048, 1), 256, 0, stream>>>(
        dd1, dw2, outp, nullptr, 3, 64, 128, 128, 128, 128, 256, 256,
        1, 2, 1, 1, 0x21, 0x21, 0x20, 0x20);
}
